// CrossModalGPG_64656437674066
// MI455X (gfx1250) — matrix operand order repaired and checked by reading, not compiled, not run on hardware
//
#include <hip/hip_runtime.h>
#include <hip/hip_bf16.h>

typedef __bf16 bf16;
typedef __attribute__((ext_vector_type(16))) __bf16        v16bf;
typedef __attribute__((ext_vector_type(8)))  float         v8f;
typedef __attribute__((ext_vector_type(8)))  unsigned int  v8u;
typedef __attribute__((ext_vector_type(4)))  unsigned int  v4u;

#define BB   8
#define CC   256
#define HW   4096
#define KK   5
#define HIDD 128
#define NTOK 32768   // BB*HW

#define ASTRIDE_Q  520   // 512 + 8 pad (bank spread)
#define ASTRIDE_W  264   // 256 + 8 pad

// ---------------------------------------------------------------------------
// Masked-pool denominators + relevance scalars.  grid = B*K, block = 256.
__global__ __launch_bounds__(256)
void kden(const float* __restrict__ mask, const float* __restrict__ rr,
          const float* __restrict__ rd, float* __restrict__ den,
          float* __restrict__ relr, float* __restrict__ reld) {
  __shared__ float s0[256], s1[256], s2[256];
  const int bid = blockIdx.x;           // = b*5 + k
  const int b = bid / KK, k = bid % KK;
  const float* mk  = mask + (size_t)(k * BB + b) * HW;
  const float* rrb = rr + (size_t)b * HW;
  const float* rdb = rd + (size_t)b * HW;
  float a0 = 0.f, a1 = 0.f, a2 = 0.f;
  for (int i = threadIdx.x; i < HW; i += 256) {
    float w = mk[i];
    a0 += w; a1 += rrb[i] * w; a2 += rdb[i] * w;
  }
  const int t = threadIdx.x;
  s0[t] = a0; s1[t] = a1; s2[t] = a2; __syncthreads();
  for (int off = 128; off > 0; off >>= 1) {
    if (t < off) { s0[t] += s0[t + off]; s1[t] += s1[t + off]; s2[t] += s2[t + off]; }
    __syncthreads();
  }
  if (t == 0) {
    float d = fmaxf(s0[0], 1e-5f);
    den[bid] = d; relr[bid] = s1[0] / d; reld[bid] = s2[0] / d;
  }
}

// ---------------------------------------------------------------------------
// Masked average pooling.  grid = B*K*C (bid = (b*5+k)*256 + c), block = 256.
__global__ __launch_bounds__(256)
void kpool(const float* __restrict__ sr, const float* __restrict__ sd,
           const float* __restrict__ mask, const float* __restrict__ den,
           float* __restrict__ pr, float* __restrict__ pd) {
  __shared__ float s0[256], s1[256];
  const int bid = blockIdx.x;
  const int c = bid & 255, bk = bid >> 8;        // bk = b*5 + k
  const int b = bk / KK, k = bk % KK;
  const size_t base  = ((size_t)(k * BB + b) * CC + c) * HW;
  const size_t mbase = (size_t)(k * BB + b) * HW;
  float ar = 0.f, ad = 0.f;
  for (int i = threadIdx.x; i < HW; i += 256) {
    float w = mask[mbase + i];
    ar += sr[base + i] * w;
    ad += sd[base + i] * w;
  }
  const int t = threadIdx.x;
  s0[t] = ar; s1[t] = ad; __syncthreads();
  for (int off = 128; off > 0; off >>= 1) {
    if (t < off) { s0[t] += s0[t + off]; s1[t] += s1[t + off]; }
    __syncthreads();
  }
  if (t == 0) {
    float d = den[bk];
    pr[bid] = s0[0] / d;
    pd[bid] = s1[0] / d;
  }
}

// p_joint = 0.5*(p_rgb + p_dep).  grid = 40, block = 256.
__global__ __launch_bounds__(256)
void kpj(const float* __restrict__ pr, const float* __restrict__ pd,
         float* __restrict__ pj) {
  int i = blockIdx.x * 256 + threadIdx.x;
  pj[i] = 0.5f * (pr[i] + pd[i]);
}

// One-shot transpose+convert: Wt[n][k] = (bf16)W[k][n].  grid = K*N/256.
__global__ __launch_bounds__(256)
void ktrans(const float* __restrict__ W, bf16* __restrict__ Wt,
            int Kdim, int Ndim) {
  int idx = blockIdx.x * 256 + threadIdx.x;
  int k = idx / Ndim, n = idx % Ndim;
  Wt[(size_t)n * Kdim + k] = (bf16)W[idx];
}

// ---------------------------------------------------------------------------
// Read a 16x32 A-fragment (bf16) from an LDS tile (row stride `stride` elems):
// VGPRs 0-3 hold K = kc+8*half .. +7, VGPRs 4-7 hold K = kc+16+8*half .. +7,
// i.e. two aligned 16-byte LDS reads.
__device__ __forceinline__
v16bf lds_afrag(const bf16* __restrict__ tile, int m, int stride, int kc, int half) {
  const bf16* ar = tile + m * stride + kc + (half << 3);
  union { v4u q[2]; v8u o; } uu;
  uu.q[0] = *(const v4u*)ar;
  uu.q[1] = *(const v4u*)(ar + 16);
  return __builtin_bit_cast(v16bf, uu.o);
}

// B-fragment from pre-transposed bf16 weight rows (pairs contiguous).
__device__ __forceinline__
v16bf g_bfrag(const bf16* __restrict__ bt, int kc, int half) {
  v8u u;
#pragma unroll
  for (int i = 0; i < 8; ++i) {
    int k0 = kc + ((i & 3) << 1) + ((i >> 2) << 4) + (half << 3);
    u[i] = *(const unsigned int*)(bt + k0);
  }
  return __builtin_bit_cast(v16bf, u);
}

// q = relu(concat(qr,qd) @ Wq_proj + bq), bf16 out [NTOK][256].
// grid = 2048 M-tiles, block = 256 (8 waves; wave -> two 16-col N-tiles).
// A tile (16 tok x 512 ch) staged once per block in LDS as bf16.
__global__ __launch_bounds__(256)
void kqproj(const float* __restrict__ fr, const float* __restrict__ fd,
            const bf16* __restrict__ WqT,   // [256][512] bf16, n-major
            const float* __restrict__ bq, bf16* __restrict__ qout) {
  __shared__ bf16 atile[16 * ASTRIDE_Q];
  const int t    = threadIdx.x;
  const int row0 = blockIdx.x * 16;
  const int b    = row0 >> 12;
  const int n0   = row0 & (HW - 1);

  // ---- cooperative A staging (coalesced: fixed k, 16 consecutive tokens) ----
  {
    const int tok = t & 15;
    const int kb  = t >> 4;             // 0..15
    const float* fr0 = fr + (size_t)b * CC * HW + n0 + tok;
    const float* fd0 = fd + (size_t)b * CC * HW + n0 + tok;
    bf16* dst = atile + tok * ASTRIDE_Q;
#pragma unroll
    for (int j = 0; j < 16; ++j) {
      int k = kb + 16 * j;
      dst[k] = (bf16)fr0[(size_t)k * HW];
    }
#pragma unroll
    for (int j = 0; j < 16; ++j) {
      int k = kb + 16 * j;
      dst[CC + k] = (bf16)fd0[(size_t)k * HW];
    }
  }
  __syncthreads();

  const int lane = t & 31;
  const int wv   = t >> 5;
  const int m    = lane & 15;
  const int half = lane >> 4;
  const int col0 = wv * 32 + m;
  const int col1 = col0 + 16;
  const bf16* bt0 = WqT + (size_t)col0 * 512;
  const bf16* bt1 = WqT + (size_t)col1 * 512;

  v8f acc0 = {}; v8f acc1 = {};
  for (int kc = 0; kc < 2 * CC; kc += 32) {
    v16bf a  = lds_afrag(atile, m, ASTRIDE_Q, kc, half);
    v16bf b0 = g_bfrag(bt0, kc, half);
    v16bf b1 = g_bfrag(bt1, kc, half);
    acc0 = __builtin_amdgcn_wmma_f32_16x16x32_bf16(false, a, false, b0, (short)0, acc0, false, false);
    acc1 = __builtin_amdgcn_wmma_f32_16x16x32_bf16(false, a, false, b1, (short)0, acc1, false, false);
  }

#pragma unroll
  for (int r = 0; r < 8; ++r) {
    int orow = row0 + r + (half << 3);
    float v0 = acc0[r] + bq[col0]; v0 = v0 > 0.f ? v0 : 0.f;
    float v1 = acc1[r] + bq[col1]; v1 = v1 > 0.f ? v1 : 0.f;
    qout[(size_t)orow * CC + col0] = (bf16)v0;
    qout[(size_t)orow * CC + col1] = (bf16)v1;
  }
}

// qWQ = q @ WQ, bf16 in/out.  grid = 2048, block = 256.  A staged in LDS.
__global__ __launch_bounds__(256)
void kqwq(const bf16* __restrict__ q, const bf16* __restrict__ WQT, // [256][256]
          bf16* __restrict__ out) {
  __shared__ bf16 atile[16 * ASTRIDE_W];
  const int t    = threadIdx.x;
  const int row0 = blockIdx.x * 16;

  // ---- cooperative A staging (u32 copies of contiguous bf16 rows) ----
  {
    const int tok = t & 15;
    const int ib  = t >> 4;             // 0..15 (u32 index base)
    const unsigned int* src = (const unsigned int*)(q + (size_t)(row0 + tok) * CC);
    unsigned int* dst = (unsigned int*)(atile + tok * ASTRIDE_W);
#pragma unroll
    for (int j = 0; j < 8; ++j) {
      int idx = ib + 16 * j;            // 0..127 u32 per row
      dst[idx] = src[idx];
    }
  }
  __syncthreads();

  const int lane = t & 31;
  const int wv   = t >> 5;
  const int m    = lane & 15;
  const int half = lane >> 4;
  const int col0 = wv * 32 + m;
  const int col1 = col0 + 16;
  const bf16* bt0 = WQT + (size_t)col0 * CC;
  const bf16* bt1 = WQT + (size_t)col1 * CC;

  v8f acc0 = {}; v8f acc1 = {};
  for (int kc = 0; kc < CC; kc += 32) {
    v16bf a  = lds_afrag(atile, m, ASTRIDE_W, kc, half);
    v16bf b0 = g_bfrag(bt0, kc, half);
    v16bf b1 = g_bfrag(bt1, kc, half);
    acc0 = __builtin_amdgcn_wmma_f32_16x16x32_bf16(false, a, false, b0, (short)0, acc0, false, false);
    acc1 = __builtin_amdgcn_wmma_f32_16x16x32_bf16(false, a, false, b1, (short)0, acc1, false, false);
  }

#pragma unroll
  for (int r = 0; r < 8; ++r) {
    int orow = row0 + r + (half << 3);
    out[(size_t)orow * CC + col0] = (bf16)acc0[r];
    out[(size_t)orow * CC + col1] = (bf16)acc1[r];
  }
}

// ---------------------------------------------------------------------------
// Small row-GEMM: out[row] = P[row] @ W  (rows = 40, dims 256x256).
__global__ __launch_bounds__(256)
void krowgemm(const float* __restrict__ P, const float* __restrict__ W,
              float* __restrict__ out) {
  __shared__ float x[256];
  const int row = blockIdx.x, t = threadIdx.x;
  x[t] = P[row * 256 + t];
  __syncthreads();
  float acc = 0.f;
  for (int i = 0; i < 256; ++i) acc += x[i] * W[i * 256 + t];
  out[row * 256 + t] = acc;
}

// Attention scores + softmax over k=5.  grid = 4096 (8 rows/block), block = 256.
__global__ __launch_bounds__(256)
void kscores(const bf16* __restrict__ qwq, const float* __restrict__ pK,
             float* __restrict__ a_out) {
  __shared__ float pk[KK * 256];
  const int row0 = blockIdx.x * 8;
  const int b = row0 >> 12;
  for (int kk = 0; kk < KK; ++kk)
    pk[kk * 256 + threadIdx.x] = pK[(b * KK + kk) * 256 + threadIdx.x];
  __syncthreads();
  const int lane = threadIdx.x & 31;
  const int row  = row0 + (threadIdx.x >> 5);
  float s[KK] = {0.f, 0.f, 0.f, 0.f, 0.f};
  for (int c = lane; c < 256; c += 32) {
    float qv = (float)qwq[(size_t)row * CC + c];
#pragma unroll
    for (int kk = 0; kk < KK; ++kk) s[kk] += qv * pk[kk * 256 + c];
  }
#pragma unroll
  for (int kk = 0; kk < KK; ++kk)
    for (int off = 16; off > 0; off >>= 1) s[kk] += __shfl_xor(s[kk], off, 32);
  if (lane == 0) {
    float mx = -1e30f;
#pragma unroll
    for (int kk = 0; kk < KK; ++kk) { s[kk] *= 0.0625f; mx = fmaxf(mx, s[kk]); }
    float sum = 0.f;
#pragma unroll
    for (int kk = 0; kk < KK; ++kk) { s[kk] = __expf(s[kk] - mx); sum += s[kk]; }
    float inv = 1.f / sum;
#pragma unroll
    for (int kk = 0; kk < KK; ++kk) a_out[(size_t)row * KK + kk] = s[kk] * inv;
  }
}

// m[b,k,c] = sum_n a[b,n,k] * q[b,n,c].  grid = 40 (b*5+k), block = 256 (c).
__global__ __launch_bounds__(256)
void km(const float* __restrict__ a, const bf16* __restrict__ q,
        float* __restrict__ mout) {
  __shared__ float av[256];
  const int bid = blockIdx.x;
  const int b = bid / KK, k = bid % KK;
  float acc = 0.f;
  for (int n0 = 0; n0 < HW; n0 += 256) {
    __syncthreads();
    av[threadIdx.x] = a[((size_t)(b * HW + n0 + threadIdx.x)) * KK + k];
    __syncthreads();
    for (int j = 0; j < 256; ++j)
      acc += av[j] * (float)q[((size_t)(b * HW + n0 + j)) * CC + threadIdx.x];
  }
  mout[bid * 256 + threadIdx.x] = acc;
}

// Generic 2-layer MLP; in_dim 513 (gated residual) or 256 (plain).  grid = 40.
__global__ __launch_bounds__(256)
void kmlp(const float* __restrict__ pin, const float* __restrict__ min_,
          const float* __restrict__ relv,
          const float* __restrict__ w1, const float* __restrict__ b1,
          const float* __restrict__ w2, const float* __restrict__ b2,
          float* __restrict__ pout, int in_dim, int gated) {
  __shared__ float x[513];
  __shared__ float h[HIDD];
  const int row = blockIdx.x, t = threadIdx.x;
  x[t] = pin[row * 256 + t];
  if (in_dim > 256) {
    x[256 + t] = min_[row * 256 + t];
    if (t == 0) x[512] = relv[row];
  }
  __syncthreads();
  if (t < HIDD) {
    float acc = b1[t];
    for (int i = 0; i < in_dim; ++i) acc += x[i] * w1[i * HIDD + t];
    h[t] = acc > 0.f ? acc : 0.f;
  }
  __syncthreads();
  float acc = b2[t];
  for (int j = 0; j < HIDD; ++j) acc += h[j] * w2[j * 256 + t];
  if (gated) pout[row * 256 + t] = pin[row * 256 + t] + acc * relv[row];
  else       pout[row * 256 + t] = acc;
}

// f_q_att[b,c,n] = sum_k a[b,n,k] * pV[b,k,c].  grid = 128 (b,chunk), block = 256 (n).
__global__ __launch_bounds__(256)
void kattout(const float* __restrict__ a, const float* __restrict__ pV,
             float* __restrict__ out) {
  __shared__ float pv[KK * 256];
  const int bid = blockIdx.x;
  const int b = bid >> 4, chunk = bid & 15;
  for (int kk = 0; kk < KK; ++kk)
    pv[kk * 256 + threadIdx.x] = pV[(b * KK + kk) * 256 + threadIdx.x];
  __syncthreads();
  const int n = chunk * 256 + threadIdx.x;
  float av[KK];
#pragma unroll
  for (int kk = 0; kk < KK; ++kk) av[kk] = a[((size_t)(b * HW + n)) * KK + kk];
  for (int c = 0; c < 256; ++c) {
    float s = av[0] * pv[c] + av[1] * pv[256 + c] + av[2] * pv[512 + c] +
              av[3] * pv[768 + c] + av[4] * pv[1024 + c];
    out[((size_t)(b * CC + c)) * HW + n] = s;
  }
}

// ---------------------------------------------------------------------------
extern "C" void kernel_launch(void* const* d_in, const int* in_sizes, int n_in,
                              void* d_out, int out_size, void* d_ws, size_t ws_size,
                              hipStream_t stream) {
  const float* f_q_rgb    = (const float*)d_in[0];
  const float* f_q_dep    = (const float*)d_in[1];
  const float* supp_rgb   = (const float*)d_in[2];
  const float* supp_dep   = (const float*)d_in[3];
  const float* supp_masks = (const float*)d_in[4];
  const float* r_rgb      = (const float*)d_in[5];
  const float* r_depth    = (const float*)d_in[6];
  const float* Wq_proj    = (const float*)d_in[7];
  const float* bq_proj    = (const float*)d_in[8];
  const float* WQ         = (const float*)d_in[9];
  const float* WK         = (const float*)d_in[10];
  const float* WV         = (const float*)d_in[11];
  const float* mrw1 = (const float*)d_in[12];
  const float* mrb1 = (const float*)d_in[13];
  const float* mrw2 = (const float*)d_in[14];
  const float* mrb2 = (const float*)d_in[15];
  const float* mdw1 = (const float*)d_in[16];
  const float* mdb1 = (const float*)d_in[17];
  const float* mdw2 = (const float*)d_in[18];
  const float* mdb2 = (const float*)d_in[19];
  const float* rrw1 = (const float*)d_in[20];
  const float* rrb1 = (const float*)d_in[21];
  const float* rrw2 = (const float*)d_in[22];
  const float* rrb2 = (const float*)d_in[23];
  const float* rdw1 = (const float*)d_in[24];
  const float* rdb1 = (const float*)d_in[25];
  const float* rdw2 = (const float*)d_in[26];
  const float* rdb2 = (const float*)d_in[27];

  char* ws = (char*)d_ws;
  bf16*  q_bf   = (bf16*)ws;                          // 16 MB
  bf16*  qwq_bf = (bf16*)(ws + (size_t)16777216);     // 16 MB
  float* f32    = (float*)(ws + (size_t)33554432);
  float* den   = f32;
  float* relr  = f32 + 64;
  float* reld  = f32 + 128;
  float* p_rgb = f32 + 192;
  float* p_dep = f32 + 10432;
  float* p_jnt = f32 + 20672;
  float* pK    = f32 + 30912;
  float* pV    = f32 + 41152;
  float* mbuf  = f32 + 51392;
  float* abuf  = f32 + 61632;                         // 32768*5 floats -> +163840
  bf16*  WqT   = (bf16*)(ws + (size_t)34456320);      // 256*512 bf16 (256 KB)
  bf16*  WQT   = (bf16*)(ws + (size_t)34718464);      // 256*256 bf16 (128 KB)

  kden <<<BB * KK, 256, 0, stream>>>(supp_masks, r_rgb, r_depth, den, relr, reld);
  kpool<<<BB * KK * CC, 256, 0, stream>>>(supp_rgb, supp_dep, supp_masks, den, p_rgb, p_dep);
  kpj  <<<BB * KK, 256, 0, stream>>>(p_rgb, p_dep, p_jnt);

  ktrans<<<(2 * CC * CC) / 256, 256, 0, stream>>>(Wq_proj, WqT, 2 * CC, CC);
  ktrans<<<(CC * CC) / 256, 256, 0, stream>>>(WQ, WQT, CC, CC);

  kqproj<<<NTOK / 16, 256, 0, stream>>>(f_q_rgb, f_q_dep, WqT, bq_proj, q_bf);
  kqwq  <<<NTOK / 16, 256, 0, stream>>>(q_bf, WQT, qwq_bf);

  for (int t = 0; t < 2; ++t) {
    krowgemm<<<BB * KK, 256, 0, stream>>>(p_jnt, WK, pK);
    kscores <<<NTOK / 8, 256, 0, stream>>>(qwq_bf, pK, abuf);
    km      <<<BB * KK, 256, 0, stream>>>(abuf, q_bf, mbuf);
    kmlp<<<BB * KK, 256, 0, stream>>>(p_rgb, mbuf, relr,
                                      mrw1 + (size_t)t * 513 * HIDD, mrb1 + (size_t)t * HIDD,
                                      mrw2 + (size_t)t * HIDD * 256, mrb2 + (size_t)t * 256,
                                      p_rgb, 513, 1);
    kmlp<<<BB * KK, 256, 0, stream>>>(p_dep, mbuf, reld,
                                      mdw1 + (size_t)t * 513 * HIDD, mdb1 + (size_t)t * HIDD,
                                      mdw2 + (size_t)t * HIDD * 256, mdb2 + (size_t)t * 256,
                                      p_dep, 513, 1);
    kpj<<<BB * KK, 256, 0, stream>>>(p_rgb, p_dep, p_jnt);
  }

  float* out = (float*)d_out;
  kmlp<<<BB * KK, 256, 0, stream>>>(p_rgb, nullptr, nullptr, rrw1, rrb1, rrw2, rrb2,
                                    out, 256, 0);
  kmlp<<<BB * KK, 256, 0, stream>>>(p_dep, nullptr, nullptr, rdw1, rdb1, rdw2, rdb2,
                                    out + 10240, 256, 0);
  krowgemm<<<BB * KK, 256, 0, stream>>>(p_jnt, WV, pV);
  kattout <<<BB * 16, 256, 0, stream>>>(abuf, pV, out + 20480);
}
